// QBMModel_12163347383026
// MI455X (gfx1250) — matrix operand order repaired and checked by reading, not compiled, not run on hardware
//
#include <hip/hip_runtime.h>
#include <stdint.h>

typedef __attribute__((ext_vector_type(16))) _Float16 v16h;
typedef __attribute__((ext_vector_type(8)))  _Float16 v8h;
typedef __attribute__((ext_vector_type(8)))  float    v8f;

#define N_NODES   64
#define N_VISIBLE 16
#define N_STEPS   20

// ---------------- RNG helpers ----------------
__device__ __forceinline__ uint32_t mix32(uint32_t h) {
    h ^= h >> 16; h *= 0x7FEB352Du;
    h ^= h >> 15; h *= 0x846CA68Bu;
    h ^= h >> 16;
    return h;
}
__device__ __forceinline__ float urand_hash(uint32_t row, uint32_t step, uint32_t node) {
    uint32_t h = row * 0x9E3779B1u ^ (step + 1u) * 0x85EBCA77u ^ (node + 1u) * 0xC2B2AE3Du;
    return (mix32(h) >> 8) * (1.0f / 16777216.0f);
}

// ---------------- Kernel 0: zero column-sum accumulator ----------------
__global__ void zero_colsum(float* colsum, int E) {
    if ((int)threadIdx.x < E) colsum[threadIdx.x] = 0.0f;
}

// ---------------- Kernel 1: column sums of embedding (HBM-bound pass) ----------------
__global__ __launch_bounds__(256) void colsum_kernel(const float* __restrict__ emb,
                                                     float* __restrict__ colsum,
                                                     int rowsPerBlock, int E, int B) {
    long r0 = (long)blockIdx.x * rowsPerBlock;
    for (int c = threadIdx.x; c < E; c += blockDim.x) {
        float acc = 0.0f;
        for (int r = 0; r < rowsPerBlock; ++r) {
            long row = r0 + r;
            if (row < B) acc += emb[row * (long)E + c];
        }
        atomicAdd(&colsum[c], acc);
    }
}

// ---------------- Kernel 2: b vector + Wsym (f16) ----------------
__global__ void prep_kernel(const float* __restrict__ colsum,
                            const float* __restrict__ pca,
                            const float* __restrict__ quad,
                            float* __restrict__ b_out,
                            _Float16* __restrict__ w_out,
                            float invB, int E) {
    int tid = threadIdx.x;
    if (tid < N_NODES) {
        float v = 0.0f;
        if (tid < N_VISIBLE) {
            for (int e = 0; e < E; ++e) v += colsum[e] * pca[e * N_VISIBLE + tid];
            v *= invB;
        }
        b_out[tid] = v;
    }
    for (int idx = tid; idx < N_NODES * N_NODES; idx += blockDim.x) {
        int i = idx >> 6, j = idx & 63;
        w_out[idx] = (_Float16)(quad[i * N_NODES + j] + quad[j * N_NODES + i]);
    }
}

// ---------------- Kernel 3: WMMA Gibbs chain + energy ----------------
// 256 threads = 8 waves; each wave owns 16 batch rows. 21 matmul passes:
// passes 0..19 sample hidden nodes, pass 20 produces z' = s @ Wsym + b.
__global__ __launch_bounds__(256) void qbm_gibbs(const float* __restrict__ reward,
                                                 const float* __restrict__ b_glob,
                                                 const _Float16* __restrict__ w_glob,
                                                 float* __restrict__ out, int B) {
    __shared__ __align__(128) _Float16 Wl[N_NODES * N_NODES]; // 8 KB, symmetric
    __shared__ float                   bl[N_NODES];
    __shared__ __align__(128) _Float16 sl[128 * N_NODES];     // 16 KB state
    __shared__ __align__(128) float    zl[128 * N_NODES];     // 32 KB z' = s @ Wsym + b

    const int tid  = threadIdx.x;
    const int wave = tid >> 5;
    const int lane = tid & 31;
    const int n    = lane & 15;   // column / row-within-tile index
    const int h    = lane >> 4;   // half-wave
    const long rowBase = (long)blockIdx.x * 128;
    const int  waveRow = wave * 16;

    for (int i = tid; i < N_NODES * N_NODES; i += 256) Wl[i] = w_glob[i];
    if (tid < N_NODES) bl[tid] = b_glob[tid];
    __syncthreads();

    // B-matrix tiles (loop-invariant). B 32x16 in this file's contiguous k order: lane n holds
    // column n, K = h*16 .. h*16+15 contiguous across 8 VGPRs. Wsym symmetric
    // => read W row (nt*16+n), cols kc*32 + h*16 .. +16, straight from LDS.
    v16h bt[4][2];
#pragma unroll
    for (int nt = 0; nt < 4; ++nt)
#pragma unroll
        for (int kc = 0; kc < 2; ++kc)
            bt[nt][kc] = *(const v16h*)(&Wl[(nt * 16 + n) * N_NODES + kc * 32 + h * 16]);

    float bvec[4];
#pragma unroll
    for (int nt = 0; nt < 4; ++nt) bvec[nt] = bl[nt * 16 + n];

    // Per-element LCG states for the 24 hidden samples this lane produces.
    // Seeded once from a hash of (global row, node) -> deterministic per call.
    uint32_t rng[3][8];
#pragma unroll
    for (int nt = 1; nt < 4; ++nt)
#pragma unroll
        for (int v = 0; v < 8; ++v) {
            long grow = rowBase + waveRow + h * 8 + v;
            uint32_t gr = (uint32_t)(grow < B ? grow : (B - 1));
            uint32_t node = (uint32_t)(nt * 16 + n);
            rng[nt - 1][v] = mix32(gr * 0x9E3779B1u ^ (node + 1u) * 0xC2B2AE3Du);
        }

    // Initialize this wave's 16 rows: visible = sigmoid(2*reward), hidden ~ Bern(0.5)
    _Float16* sw = &sl[waveRow * N_NODES];
    float*    zw = &zl[waveRow * N_NODES];
    for (int idx = lane; idx < 16 * N_NODES; idx += 32) {
        int r = idx >> 6, c = idx & 63;
        long grow = rowBase + waveRow + r;
        uint32_t gr = (uint32_t)(grow < B ? grow : (B - 1));
        float v;
        if (c < N_VISIBLE) v = __builtin_amdgcn_rcpf(1.0f + __expf(-2.0f * reward[gr]));
        else               v = (urand_hash(gr, 0x7FFFu, (uint32_t)c) < 0.5f) ? 1.0f : 0.0f;
        sw[idx] = (_Float16)v;
    }

    for (int step = 0; step <= N_STEPS; ++step) {
        // Bias folded into accumulator init: acc element (M, N=n+nt*16) = b[node].
        v8f acc[4];
#pragma unroll
        for (int nt = 0; nt < 4; ++nt)
#pragma unroll
            for (int v = 0; v < 8; ++v) acc[nt][v] = bvec[nt];

#pragma unroll
        for (int kc = 0; kc < 2; ++kc) {
            // A 16x32 in the same contiguous k order as B: lane holds row n; K = h*16 .. h*16+15 contiguous
            union { v16h v; v8h p[2]; } A;
            const _Float16* srow = sw + n * N_NODES + kc * 32;
            A.p[0] = *(const v8h*)(srow + h * 16);
            A.p[1] = *(const v8h*)(srow + h * 16 + 8);
            acc[0] = __builtin_amdgcn_wmma_f32_16x16x32_f16(false, A.v, false, bt[0][kc], (short)0, acc[0], false, false);
            acc[1] = __builtin_amdgcn_wmma_f32_16x16x32_f16(false, A.v, false, bt[1][kc], (short)0, acc[1], false, false);
            acc[2] = __builtin_amdgcn_wmma_f32_16x16x32_f16(false, A.v, false, bt[2][kc], (short)0, acc[2], false, false);
            acc[3] = __builtin_amdgcn_wmma_f32_16x16x32_f16(false, A.v, false, bt[3][kc], (short)0, acc[3], false, false);
        }

        if (step < N_STEPS) {
            // Sample hidden nodes (n-tiles 1..3). Visible tile 0 stays clamped.
            // C layout: lane (n,h), VGPR v -> element (M = h*8+v, N = n).
            // Bernoulli test without division:  u < 1/(1+e)  <=>  u + u*e < 1.
#pragma unroll
            for (int nt = 1; nt < 4; ++nt) {
#pragma unroll
                for (int v = 0; v < 8; ++v) {
                    int M = h * 8 + v;
                    int node = nt * 16 + n;
                    uint32_t r = rng[nt - 1][v] * 1664525u + 1013904223u;
                    rng[nt - 1][v] = r;
                    float u = (r >> 8) * (1.0f / 16777216.0f);
                    float e = __expf(-acc[nt][v]);          // acc already includes b
                    float s = (fmaf(u, e, u) < 1.0f) ? 1.0f : 0.0f;
                    sw[M * N_NODES + node] = (_Float16)s;
                }
            }
        } else {
            // Final pass: stash z' = s @ Wsym + b for the energy reduction.
#pragma unroll
            for (int nt = 0; nt < 4; ++nt)
#pragma unroll
                for (int v = 0; v < 8; ++v) {
                    int M = h * 8 + v;
                    zw[M * N_NODES + nt * 16 + n] = acc[nt][v];
                }
        }
    }

    __syncthreads();

    // Energy finish. With z' = Wsym s + b:
    //   s.z' = s^T Wsym s + s.b = 2 s^T Q s + s.b
    //   adjusted = reward + s^T Q s + s.b = reward + 0.5*(s.z') + 0.5*(s.b)
    if (tid < 128) {
        long grow = rowBase + tid;
        if (grow < B) {
            const _Float16* srow = &sl[tid * N_NODES];
            const float*    zrow = &zl[tid * N_NODES];
            float dot = 0.0f, sb = 0.0f;
#pragma unroll
            for (int j = 0; j < N_NODES; ++j) {
                float sj = (float)srow[j];
                dot += sj * zrow[j];
                sb  += sj * bl[j];
            }
            out[grow] = reward[grow] + 0.5f * dot + 0.5f * sb;
        }
    }
}

// ---------------- Host launcher ----------------
extern "C" void kernel_launch(void* const* d_in, const int* in_sizes, int n_in,
                              void* d_out, int out_size, void* d_ws, size_t ws_size,
                              hipStream_t stream) {
    const float* emb    = (const float*)d_in[0];  // (B, E) f32
    const float* reward = (const float*)d_in[1];  // (B,)   f32
    const float* pca    = (const float*)d_in[2];  // (E,16) f32
    const float* quad   = (const float*)d_in[3];  // (64,64) f32
    float* out = (float*)d_out;                   // (B,1)  f32

    const int B = in_sizes[1];
    const int E = in_sizes[0] / B;

    float*    colsum = (float*)d_ws;                            // E floats
    float*    b_ws   = (float*)((char*)d_ws + 2048);            // 64 floats
    _Float16* w_ws   = (_Float16*)((char*)d_ws + 4096);         // 64*64 f16

    zero_colsum<<<1, (E + 63) & ~63, 0, stream>>>(colsum, E);

    const int nb1 = 512;
    const int rpb = (B + nb1 - 1) / nb1;
    colsum_kernel<<<nb1, 256, 0, stream>>>(emb, colsum, rpb, E, B);

    prep_kernel<<<1, 256, 0, stream>>>(colsum, pca, quad, b_ws, w_ws, 1.0f / (float)B, E);

    const int nb3 = (B + 127) / 128;
    qbm_gibbs<<<nb3, 256, 0, stream>>>(reward, b_ws, w_ws, out, B);
}
